// EnhancedBidirectionalCrossAttention_62397284876958
// MI455X (gfx1250) — matrix operand order repaired and checked by reading, not compiled, not run on hardware
//
#include <hip/hip_runtime.h>
#include <hip/hip_bf16.h>

#define DEV __device__ __forceinline__

typedef __attribute__((ext_vector_type(16))) __bf16 v16bf;
typedef __attribute__((ext_vector_type(8)))  float  v8f;

constexpr int D   = 512;
constexpr int FF  = 2048;
constexpr int NB  = 8;      // batch
constexpr int SI  = 4096;   // img seq
constexpr int STX = 77;     // text seq
constexpr int NH  = 8;      // heads
constexpr int HD  = 64;     // head dim
constexpr int NI  = NB * SI;   // 32768 img tokens
constexpr int NT  = NB * STX;  // 616 text tokens
constexpr float ASCALE = 0.125f; // 1/sqrt(64)

// ---------------- helpers ----------------

union U128 { uint4 u; __bf16 h[8]; };

DEV U128 ld128(const __bf16* p) { U128 r; r.u = *reinterpret_cast<const uint4*>(p); return r; }

DEV __bf16 f2bf(float x) {
  unsigned u = __float_as_uint(x);
  unsigned r = (u + 0x7FFFu + ((u >> 16) & 1u)) >> 16;  // RNE
  unsigned short s = (unsigned short)r;
  __bf16 out;
  __builtin_memcpy(&out, &s, 2);
  return out;
}

DEV v8f wmma_bf16(v16bf a, v16bf b, v8f c) {
  return __builtin_amdgcn_wmma_f32_16x16x32_bf16(false, a, false, b, (short)0, c, false, false);
}

// CDNA5 async copy: global -> LDS, 16 bytes per active lane, tracked by ASYNCcnt.
// Generic pointers to LDS carry the LDS byte offset in their low 32 bits (ISA 10.2).
DEV void async_copy_b128(const __bf16* gptr, __bf16* lptr) {
  unsigned lds = (unsigned)(uintptr_t)lptr;
  asm volatile("global_load_async_to_lds_b128 %0, %1, off"
               :: "v"(lds), "v"(gptr) : "memory");
}
DEV void wait_async0() { asm volatile("s_wait_asynccnt 0" ::: "memory"); }

// A fragment: 16 rows x 32 K (bf16), row-major source with leading dim ld.
// Lane L<16: elems 0..7 = K(k0+0..7), 8..15 = K(k0+16..23); L>=16: +8 / +24.
DEV v16bf frag_a(const __bf16* p0, int ld) {
  int lane = threadIdx.x & 31;
  const __bf16* p = p0 + (size_t)(lane & 15) * ld;
  U128 lo = ld128(p + ((lane < 16) ? 0 : 8));
  U128 hi = ld128(p + ((lane < 16) ? 16 : 24));
  v16bf a;
#pragma unroll
  for (int i = 0; i < 8; i++) { a[i] = lo.h[i]; a[i + 8] = hi.h[i]; }
  return a;
}

DEV v16bf frag_a_clamp(const __bf16* mat, int row0, int M, int k0, int ld) {
  int lane = threadIdx.x & 31;
  int r = row0 + (lane & 15); if (r > M - 1) r = M - 1;
  const __bf16* p = mat + (size_t)r * ld + k0;
  U128 lo = ld128(p + ((lane < 16) ? 0 : 8));
  U128 hi = ld128(p + ((lane < 16) ? 16 : 24));
  v16bf a;
#pragma unroll
  for (int i = 0; i < 8; i++) { a[i] = lo.h[i]; a[i + 8] = hi.h[i]; }
  return a;
}

// B fragment: 32 K x 16 N, source stored rows = N (i.e. W[n][k]), leading dim ld.
// Lane L<16: column n=L, elems 0..7 = K(k0+0..7), 8..15 = K(k0+16..23); L>=16: +8 / +24.
DEV v16bf frag_b(const __bf16* p0, int ld) {
  int lane = threadIdx.x & 31;
  const __bf16* p = p0 + (size_t)(lane & 15) * ld;
  U128 lo = ld128(p + ((lane < 16) ? 0 : 8));
  U128 hi = ld128(p + ((lane < 16) ? 16 : 24));
  v16bf b;
#pragma unroll
  for (int i = 0; i < 8; i++) { b[i] = lo.h[i]; b[i + 8] = hi.h[i]; }
  return b;
}

DEV v8f zero8() { v8f z;
#pragma unroll
  for (int i = 0; i < 8; i++) z[i] = 0.f; return z; }

// ---------------- conversion ----------------

__global__ void cvt_bf16(const float* __restrict__ in, __bf16* __restrict__ out, size_t n) {
  size_t i = (size_t)blockIdx.x * blockDim.x + threadIdx.x;
  size_t stride = (size_t)gridDim.x * blockDim.x;
  for (; i < n; i += stride) out[i] = f2bf(in[i]);
}

// ---------------- GEMM: C = A[MxK] * W[NxK]^T + bias ----------------
// 256 threads = 8 waves; block tile 128(M) x 64(N); K step 32.
// B tile (64x32 = 4KB) double-buffered in LDS via async global->LDS copies.
// EPI: 0 -> f32 out, 1 -> bf16 out, 2 -> gelu(out) as bf16

template<int EPI>
__global__ __launch_bounds__(256)
void gemm_bf16(const __bf16* __restrict__ A, const __bf16* __restrict__ W,
               const float* __restrict__ bias, float* __restrict__ Cf,
               __bf16* __restrict__ Cb, int M, int N, int K) {
  __shared__ __align__(16) __bf16 Bl[2][64][32];

  int wave = threadIdx.x >> 5;
  int lane = threadIdx.x & 31;
  int row0 = blockIdx.x * 128 + wave * 16;
  int col0 = blockIdx.y * 64;

  // per-thread staging slot: thread t loads W[col0 + t/4][k0 + (t%4)*8 .. +8)
  int sn = threadIdx.x >> 2;
  int sk = (threadIdx.x & 3) * 8;

  v8f acc[4];
#pragma unroll
  for (int t = 0; t < 4; t++) acc[t] = zero8();

  // prologue: stage k0 = 0 into buffer 0
  async_copy_b128(W + (size_t)(col0 + sn) * K + sk, &Bl[0][sn][sk]);
  wait_async0();
  __syncthreads();

  int buf = 0;
  for (int k0 = 0; k0 < K; k0 += 32) {
    if (k0 + 32 < K)  // issue next chunk into the other buffer (async, overlap w/ math)
      async_copy_b128(W + (size_t)(col0 + sn) * K + (k0 + 32) + sk, &Bl[buf ^ 1][sn][sk]);

    v16bf a = frag_a_clamp(A, row0, M, k0, K);
#pragma unroll
    for (int t = 0; t < 4; t++) {
      v16bf b = frag_b(&Bl[buf][t * 16][0], 32);
      acc[t] = wmma_bf16(a, b, acc[t]);
    }

    wait_async0();     // own async loads landed in LDS
    __syncthreads();   // everyone's landed; current buffer fully consumed
    buf ^= 1;
  }

  int rbase = row0 + ((lane < 16) ? 0 : 8);
  int c = lane & 15;
#pragma unroll
  for (int t = 0; t < 4; t++) {
    int col = col0 + t * 16 + c;
    float bv = bias ? bias[col] : 0.f;
#pragma unroll
    for (int r = 0; r < 8; r++) {
      int row = rbase + r;
      if (row < M) {
        float y = acc[t][r] + bv;
        if (EPI == 2) y = 0.5f * y * (1.f + erff(y * 0.70710678118f)); // exact gelu
        size_t idx = (size_t)row * N + col;
        if (EPI == 0) Cf[idx] = y; else Cb[idx] = f2bf(y);
      }
    }
  }
}

// ---------------- LayerNorm family (one block of 128 per row, D=512) ----------------

DEV float block_sum128(float v, float* red) {
  int t = threadIdx.x;
  red[t] = v; __syncthreads();
#pragma unroll
  for (int s = 64; s > 0; s >>= 1) { if (t < s) red[t] += red[t + s]; __syncthreads(); }
  float r = red[0]; __syncthreads();
  return r;
}

__global__ __launch_bounds__(128)
void ln_fwd(const float* __restrict__ x, const float* __restrict__ g, const float* __restrict__ b,
            float* __restrict__ yf, __bf16* __restrict__ yb, int relu) {
  __shared__ float red[128];
  size_t row = blockIdx.x;
  const float* xr = x + row * (size_t)D;
  float loc[4]; float s = 0.f;
#pragma unroll
  for (int i = 0; i < 4; i++) { loc[i] = xr[threadIdx.x + i * 128]; s += loc[i]; }
  float mean = block_sum128(s, red) * (1.f / D);
  float vs = 0.f;
#pragma unroll
  for (int i = 0; i < 4; i++) { float d0 = loc[i] - mean; vs += d0 * d0; }
  float inv = rsqrtf(block_sum128(vs, red) * (1.f / D) + 1e-5f);
#pragma unroll
  for (int i = 0; i < 4; i++) {
    int c = threadIdx.x + i * 128;
    float y = (loc[i] - mean) * inv * g[c] + b[c];
    if (relu) y = fmaxf(y, 0.f);
    if (yf) yf[row * D + c] = y;
    if (yb) yb[row * D + c] = f2bf(y);
  }
}

__global__ __launch_bounds__(128)
void ln_gate(const float* __restrict__ x, const float* __restrict__ g, const float* __restrict__ b,
             const float* __restrict__ wg, __bf16* __restrict__ yb, float* __restrict__ gate) {
  __shared__ float red[128];
  size_t row = blockIdx.x;
  const float* xr = x + row * (size_t)D;
  float loc[4]; float s = 0.f;
#pragma unroll
  for (int i = 0; i < 4; i++) { loc[i] = xr[threadIdx.x + i * 128]; s += loc[i]; }
  float mean = block_sum128(s, red) * (1.f / D);
  float vs = 0.f;
#pragma unroll
  for (int i = 0; i < 4; i++) { float d0 = loc[i] - mean; vs += d0 * d0; }
  float inv = rsqrtf(block_sum128(vs, red) * (1.f / D) + 1e-5f);
  float acc = 0.f;
#pragma unroll
  for (int i = 0; i < 4; i++) {
    int c = threadIdx.x + i * 128;
    float y = (loc[i] - mean) * inv * g[c] + b[c];
    yb[row * D + c] = f2bf(y);
    acc += y * wg[c];
  }
  float dot = block_sum128(acc, red);
  if (threadIdx.x == 0) gate[row] = 1.f / (1.f + __expf(-dot));
}

// out = resid + LN(gate[row]*o) * g + b
__global__ __launch_bounds__(128)
void ln_gated_res(const float* __restrict__ o, const float* __restrict__ gate,
                  const float* __restrict__ g, const float* __restrict__ b,
                  const float* __restrict__ resid, float* __restrict__ out) {
  __shared__ float red[128];
  size_t row = blockIdx.x;
  float gg = gate[row];
  const float* xr = o + row * (size_t)D;
  float loc[4]; float s = 0.f;
#pragma unroll
  for (int i = 0; i < 4; i++) { loc[i] = gg * xr[threadIdx.x + i * 128]; s += loc[i]; }
  float mean = block_sum128(s, red) * (1.f / D);
  float vs = 0.f;
#pragma unroll
  for (int i = 0; i < 4; i++) { float d0 = loc[i] - mean; vs += d0 * d0; }
  float inv = rsqrtf(block_sum128(vs, red) * (1.f / D) + 1e-5f);
#pragma unroll
  for (int i = 0; i < 4; i++) {
    int c = threadIdx.x + i * 128;
    float y = (loc[i] - mean) * inv * g[c] + b[c];
    out[row * D + c] = resid[row * D + c] + y;
  }
}

// out = LN(a + f) * g + b
__global__ __launch_bounds__(128)
void ln_add(const float* __restrict__ a, const float* __restrict__ f,
            const float* __restrict__ g, const float* __restrict__ b,
            float* __restrict__ out) {
  __shared__ float red[128];
  size_t row = blockIdx.x;
  float loc[4]; float s = 0.f;
#pragma unroll
  for (int i = 0; i < 4; i++) {
    int c = threadIdx.x + i * 128;
    loc[i] = a[row * D + c] + f[row * D + c]; s += loc[i];
  }
  float mean = block_sum128(s, red) * (1.f / D);
  float vs = 0.f;
#pragma unroll
  for (int i = 0; i < 4; i++) { float d0 = loc[i] - mean; vs += d0 * d0; }
  float inv = rsqrtf(block_sum128(vs, red) * (1.f / D) + 1e-5f);
#pragma unroll
  for (int i = 0; i < 4; i++) {
    int c = threadIdx.x + i * 128;
    out[row * D + c] = (loc[i] - mean) * inv * g[c] + b[c];
  }
}

// ---------------- t2i attention: 32768 queries x 77 keys ----------------
// grid (SI/64, NH, NB), 128 threads (4 waves); keys padded 77->96

__global__ __launch_bounds__(128)
void attn_t2i(const __bf16* __restrict__ q, const __bf16* __restrict__ k,
              const __bf16* __restrict__ v, __bf16* __restrict__ out) {
  __shared__ __align__(16) __bf16 Kl[96][64];
  __shared__ __align__(16) __bf16 Vt[64][96];
  __shared__ __align__(16) float  Sl[4][16][96];
  __shared__ __align__(16) __bf16 Pl[4][16][96];

  int b = blockIdx.z, h = blockIdx.y;
  int tid = threadIdx.x, wave = tid >> 5, lane = tid & 31;

  for (int i = tid; i < 96 * 64; i += 128) {
    int kk = i >> 6, d = i & 63;
    __bf16 zk = f2bf(0.f), zv = zk;
    if (kk < STX) {
      size_t base = ((size_t)(b * STX + kk)) * D + h * HD + d;
      zk = k[base]; zv = v[base];
    }
    Kl[kk][d] = zk; Vt[d][kk] = zv;
  }
  __syncthreads();

  size_t qrow0 = (size_t)b * SI + (size_t)blockIdx.x * 64 + wave * 16;
  const __bf16* qbase = q + qrow0 * D + h * HD;
  int rb = (lane < 16) ? 0 : 8, c = lane & 15;

  // S = Q K^T * scale
#pragma unroll
  for (int t = 0; t < 6; t++) {
    v8f acc = zero8();
#pragma unroll
    for (int k0 = 0; k0 < 64; k0 += 32) {
      v16bf a  = frag_a(qbase + k0, D);
      v16bf bb = frag_b(&Kl[t * 16][0] + k0, 64);
      acc = wmma_bf16(a, bb, acc);
    }
#pragma unroll
    for (int r = 0; r < 8; r++) Sl[wave][rb + r][t * 16 + c] = acc[r] * ASCALE;
  }
  __syncthreads();

  // masked softmax per row
  if (tid < 64) {
    int w = tid >> 4, r = tid & 15;
    float mx = -3e38f;
    for (int j = 0; j < STX; j++) mx = fmaxf(mx, Sl[w][r][j]);
    float ssum = 0.f;
    for (int j = 0; j < 96; j++) {
      float e = (j < STX) ? __expf(Sl[w][r][j] - mx) : 0.f;
      ssum += e; Sl[w][r][j] = e;
    }
    float invs = 1.f / ssum;
    for (int j = 0; j < 96; j++) Pl[w][r][j] = f2bf(Sl[w][r][j] * invs);
  }
  __syncthreads();

  // O = P V
#pragma unroll
  for (int t = 0; t < 4; t++) {
    v8f acc = zero8();
#pragma unroll
    for (int k0 = 0; k0 < 96; k0 += 32) {
      v16bf a  = frag_a(&Pl[wave][0][0] + k0, 96);
      v16bf bb = frag_b(&Vt[t * 16][0] + k0, 96);
      acc = wmma_bf16(a, bb, acc);
    }
#pragma unroll
    for (int r = 0; r < 8; r++) {
      size_t row = qrow0 + rb + r;
      out[row * D + h * HD + t * 16 + c] = f2bf(acc[r]);
    }
  }
}

// ---------------- i2t attention (flash): 77 queries x 4096 keys ----------------
// grid (NB*NH), 160 threads (5 waves), queries padded 77->80, 64-key chunks

__global__ __launch_bounds__(160)
void attn_i2t(const __bf16* __restrict__ q, const __bf16* __restrict__ k,
              const __bf16* __restrict__ v, __bf16* __restrict__ out) {
  __shared__ __align__(16) __bf16 Ql[80][64];
  __shared__ __align__(16) float  Ol[80][64];
  __shared__ __align__(16) __bf16 Kl[64][64];
  __shared__ __align__(16) __bf16 Vt[64][64];
  __shared__ __align__(16) float  Sl[5][16][64];
  __shared__ __align__(16) __bf16 Pl[5][16][64];
  __shared__ float mrow[80], lrow[80], arow[80];

  int b = blockIdx.x >> 3, h = blockIdx.x & 7;
  int tid = threadIdx.x, wave = tid >> 5, lane = tid & 31;
  int rb = (lane < 16) ? 0 : 8, c = lane & 15;

  for (int i = tid; i < 80 * 64; i += 160) {
    int r = i >> 6, d = i & 63;
    Ol[r][d] = 0.f;
    Ql[r][d] = (r < STX) ? q[((size_t)(b * STX + r)) * D + h * HD + d] : f2bf(0.f);
  }
  for (int i = tid; i < 80; i += 160) { mrow[i] = -3e38f; lrow[i] = 0.f; arow[i] = 0.f; }
  __syncthreads();

  for (int c0 = 0; c0 < SI; c0 += 64) {
    for (int i = tid; i < 64 * 64; i += 160) {
      int r = i >> 6, d = i & 63;
      size_t base = ((size_t)(b * SI + c0 + r)) * D + h * HD + d;
      Kl[r][d] = k[base];
      Vt[d][r] = v[base];
    }
    __syncthreads();

    // S chunk
#pragma unroll
    for (int t = 0; t < 4; t++) {
      v8f acc = zero8();
#pragma unroll
      for (int k0 = 0; k0 < 64; k0 += 32) {
        v16bf a  = frag_a(&Ql[wave * 16][0] + k0, 64);
        v16bf bb = frag_b(&Kl[t * 16][0] + k0, 64);
        acc = wmma_bf16(a, bb, acc);
      }
#pragma unroll
      for (int r = 0; r < 8; r++) Sl[wave][rb + r][t * 16 + c] = acc[r] * ASCALE;
    }
    __syncthreads();

    // online softmax bookkeeping
    if (tid < 80) {
      int w = tid >> 4, r = tid & 15;
      float mc = -3e38f;
      for (int j = 0; j < 64; j++) mc = fmaxf(mc, Sl[w][r][j]);
      float mnew = fmaxf(mrow[tid], mc);
      float al = __expf(mrow[tid] - mnew);
      float ss = 0.f;
      for (int j = 0; j < 64; j++) {
        float e = __expf(Sl[w][r][j] - mnew);
        ss += e; Pl[w][r][j] = f2bf(e);
      }
      lrow[tid] = lrow[tid] * al + ss;
      mrow[tid] = mnew;
      arow[tid] = al;
    }
    __syncthreads();

    for (int i = tid; i < 80 * 64; i += 160) { int r = i >> 6, d = i & 63; Ol[r][d] *= arow[r]; }
    __syncthreads();

    // O += P V
#pragma unroll
    for (int t = 0; t < 4; t++) {
      v8f acc = zero8();
#pragma unroll
      for (int k0 = 0; k0 < 64; k0 += 32) {
        v16bf a  = frag_a(&Pl[wave][0][0] + k0, 64);
        v16bf bb = frag_b(&Vt[t * 16][0] + k0, 64);
        acc = wmma_bf16(a, bb, acc);
      }
#pragma unroll
      for (int r = 0; r < 8; r++) Ol[wave * 16 + rb + r][t * 16 + c] += acc[r];
    }
    __syncthreads();
  }

  for (int i = tid; i < 80 * 64; i += 160) {
    int r = i >> 6, d = i & 63;
    if (r < STX) out[((size_t)(b * STX + r)) * D + h * HD + d] = f2bf(Ol[r][d] / lrow[r]);
  }
}

// ---------------- host orchestration ----------------

extern "C" void kernel_launch(void* const* d_in, const int* in_sizes, int n_in,
                              void* d_out, int out_size, void* d_ws, size_t ws_size,
                              hipStream_t stream) {
  (void)in_sizes; (void)n_in; (void)out_size; (void)ws_size;

  enum { I_IMG = 0, I_TXT = 1, T2I = 2, I2T = 17,
         IMG_ENH_W = 32, IMG_ENH_B, IMG_ENH_G, IMG_ENH_BN,
         TXT_ENH_W, TXT_ENH_B, TXT_ENH_G, TXT_ENH_BN,
         IMG_N1_G, IMG_N1_B, IMG_FF_W1, IMG_FF_B1, IMG_FF_W2, IMG_FF_B2, IMG_N2_G, IMG_N2_B,
         TXT_N1_G, TXT_N1_B, TXT_FF_W1, TXT_FF_B1, TXT_FF_W2, TXT_FF_B2, TXT_N2_G, TXT_N2_B };
  enum { A_NQ_G = 0, A_NQ_B, A_NKV_G, A_NKV_B, A_WQ, A_BQ, A_WK, A_BK,
         A_WV, A_BV, A_WO, A_BO, A_WG, A_NO_G, A_NO_B };

  auto FP = [&](int i) { return (const float*)d_in[i]; };

  char* wsb = (char*)d_ws; size_t off = 0;
  auto alloc = [&](size_t bytes) { void* p = wsb + off; off = (off + bytes + 255) & ~(size_t)255; return p; };

  auto cvtw = [&](int idx, size_t n) {
    __bf16* dst = (__bf16*)alloc(n * 2);
    int blocks = (int)((n + 1023) / 1024); if (blocks > 2048) blocks = 2048;
    cvt_bf16<<<blocks, 256, 0, stream>>>(FP(idx), dst, n);
    return (const __bf16*)dst;
  };

  // weights -> bf16
  const __bf16* w_img_enh = cvtw(IMG_ENH_W, (size_t)D * D);
  const __bf16* w_txt_enh = cvtw(TXT_ENH_W, (size_t)D * D);
  const __bf16* t2i_wq = cvtw(T2I + A_WQ, (size_t)D * D);
  const __bf16* t2i_wk = cvtw(T2I + A_WK, (size_t)D * D);
  const __bf16* t2i_wv = cvtw(T2I + A_WV, (size_t)D * D);
  const __bf16* t2i_wo = cvtw(T2I + A_WO, (size_t)D * D);
  const __bf16* i2t_wq = cvtw(I2T + A_WQ, (size_t)D * D);
  const __bf16* i2t_wk = cvtw(I2T + A_WK, (size_t)D * D);
  const __bf16* i2t_wv = cvtw(I2T + A_WV, (size_t)D * D);
  const __bf16* i2t_wo = cvtw(I2T + A_WO, (size_t)D * D);
  const __bf16* img_w1 = cvtw(IMG_FF_W1, (size_t)FF * D);
  const __bf16* img_w2 = cvtw(IMG_FF_W2, (size_t)D * FF);
  const __bf16* txt_w1 = cvtw(TXT_FF_W1, (size_t)FF * D);
  const __bf16* txt_w2 = cvtw(TXT_FF_W2, (size_t)D * FF);

  // activations / scratch
  __bf16* x_img_bf = (__bf16*)alloc((size_t)NI * D * 2);
  __bf16* x_txt_bf = (__bf16*)alloc((size_t)NT * D * 2);
  float*  y_img_f  = (float*)alloc((size_t)NI * D * 4);
  float*  y_txt_f  = (float*)alloc((size_t)NT * D * 4);
  float*  img_e_f  = (float*)alloc((size_t)NI * D * 4);
  float*  txt_e_f  = (float*)alloc((size_t)NT * D * 4);
  float*  enh_img  = (float*)alloc((size_t)NI * D * 4);
  float*  enh_txt  = (float*)alloc((size_t)NT * D * 4);
  __bf16* q_img_bf = (__bf16*)alloc((size_t)NI * D * 2);
  __bf16* k_txt_bf = (__bf16*)alloc((size_t)NT * D * 2);
  __bf16* v_txt_bf = (__bf16*)alloc((size_t)NT * D * 2);
  __bf16* ao_img   = (__bf16*)alloc((size_t)NI * D * 2);
  __bf16* q_txt_bf = (__bf16*)alloc((size_t)NT * D * 2);
  __bf16* k_img_bf = (__bf16*)alloc((size_t)NI * D * 2);
  __bf16* v_img_bf = (__bf16*)alloc((size_t)NI * D * 2);
  __bf16* ao_txt   = (__bf16*)alloc((size_t)NT * D * 2);
  float*  gate_img = (float*)alloc((size_t)NI * 4);
  float*  gate_txt = (float*)alloc((size_t)NT * 4);
  __bf16* h_img_bf = (__bf16*)alloc((size_t)NI * FF * 2);
  __bf16* h_txt_bf = (__bf16*)alloc((size_t)NT * FF * 2);

  auto gemm = [&](int epi, const __bf16* A, const __bf16* W, const float* bias,
                  float* Cf, __bf16* Cb, int M, int N, int K) {
    dim3 grid((M + 127) / 128, N / 64);
    if (epi == 0)      gemm_bf16<0><<<grid, 256, 0, stream>>>(A, W, bias, Cf, Cb, M, N, K);
    else if (epi == 1) gemm_bf16<1><<<grid, 256, 0, stream>>>(A, W, bias, Cf, Cb, M, N, K);
    else               gemm_bf16<2><<<grid, 256, 0, stream>>>(A, W, bias, Cf, Cb, M, N, K);
  };

  // ---- Stage A: feature enhancement ----
  cvt_bf16<<<2048, 256, 0, stream>>>(FP(I_IMG), x_img_bf, (size_t)NI * D);
  cvt_bf16<<<256, 256, 0, stream>>>(FP(I_TXT), x_txt_bf, (size_t)NT * D);
  gemm(0, x_img_bf, w_img_enh, FP(IMG_ENH_B), y_img_f, nullptr, NI, D, D);
  gemm(0, x_txt_bf, w_txt_enh, FP(TXT_ENH_B), y_txt_f, nullptr, NT, D, D);
  ln_fwd<<<NI, 128, 0, stream>>>(y_img_f, FP(IMG_ENH_G), FP(IMG_ENH_BN), img_e_f, nullptr, 1);
  ln_fwd<<<NT, 128, 0, stream>>>(y_txt_f, FP(TXT_ENH_G), FP(TXT_ENH_BN), txt_e_f, nullptr, 1);

  // ---- Stage B: t2i MHA (q = img_e, kv = txt_e) ----
  ln_gate<<<NI, 128, 0, stream>>>(img_e_f, FP(T2I + A_NQ_G), FP(T2I + A_NQ_B), FP(T2I + A_WG), x_img_bf, gate_img);
  ln_fwd<<<NT, 128, 0, stream>>>(txt_e_f, FP(T2I + A_NKV_G), FP(T2I + A_NKV_B), nullptr, x_txt_bf, 0);
  gemm(1, x_img_bf, t2i_wq, FP(T2I + A_BQ), nullptr, q_img_bf, NI, D, D);
  gemm(1, x_txt_bf, t2i_wk, FP(T2I + A_BK), nullptr, k_txt_bf, NT, D, D);
  gemm(1, x_txt_bf, t2i_wv, FP(T2I + A_BV), nullptr, v_txt_bf, NT, D, D);
  attn_t2i<<<dim3(SI / 64, NH, NB), 128, 0, stream>>>(q_img_bf, k_txt_bf, v_txt_bf, ao_img);
  gemm(0, ao_img, t2i_wo, FP(T2I + A_BO), y_img_f, nullptr, NI, D, D);
  ln_gated_res<<<NI, 128, 0, stream>>>(y_img_f, gate_img, FP(T2I + A_NO_G), FP(T2I + A_NO_B), img_e_f, enh_img);

  // ---- Stage C: i2t MHA (q = txt_e, kv = img_e) ----
  ln_gate<<<NT, 128, 0, stream>>>(txt_e_f, FP(I2T + A_NQ_G), FP(I2T + A_NQ_B), FP(I2T + A_WG), x_txt_bf, gate_txt);
  ln_fwd<<<NI, 128, 0, stream>>>(img_e_f, FP(I2T + A_NKV_G), FP(I2T + A_NKV_B), nullptr, x_img_bf, 0);
  gemm(1, x_txt_bf, i2t_wq, FP(I2T + A_BQ), nullptr, q_txt_bf, NT, D, D);
  gemm(1, x_img_bf, i2t_wk, FP(I2T + A_BK), nullptr, k_img_bf, NI, D, D);
  gemm(1, x_img_bf, i2t_wv, FP(I2T + A_BV), nullptr, v_img_bf, NI, D, D);
  attn_i2t<<<NB * NH, 160, 0, stream>>>(q_txt_bf, k_img_bf, v_img_bf, ao_txt);
  gemm(0, ao_txt, i2t_wo, FP(I2T + A_BO), y_txt_f, nullptr, NT, D, D);
  ln_gated_res<<<NT, 128, 0, stream>>>(y_txt_f, gate_txt, FP(I2T + A_NO_G), FP(I2T + A_NO_B), txt_e_f, enh_txt);

  // ---- Stage D: FFN + final LN ----
  float* out_img = (float*)d_out;
  float* out_txt = (float*)d_out + (size_t)NI * D;

  ln_fwd<<<NI, 128, 0, stream>>>(enh_img, FP(IMG_N1_G), FP(IMG_N1_B), nullptr, x_img_bf, 0);
  gemm(2, x_img_bf, img_w1, FP(IMG_FF_B1), nullptr, h_img_bf, NI, FF, D);
  gemm(0, h_img_bf, img_w2, FP(IMG_FF_B2), y_img_f, nullptr, NI, D, FF);
  ln_add<<<NI, 128, 0, stream>>>(enh_img, y_img_f, FP(IMG_N2_G), FP(IMG_N2_B), out_img);

  ln_fwd<<<NT, 128, 0, stream>>>(enh_txt, FP(TXT_N1_G), FP(TXT_N1_B), nullptr, x_txt_bf, 0);
  gemm(2, x_txt_bf, txt_w1, FP(TXT_FF_B1), nullptr, h_txt_bf, NT, FF, D);
  gemm(0, h_txt_bf, txt_w2, FP(TXT_FF_B2), y_txt_f, nullptr, NT, D, FF);
  ln_add<<<NT, 128, 0, stream>>>(enh_txt, y_txt_f, FP(TXT_N2_G), FP(TXT_N2_B), out_txt);
}
